// TextGuideAttentionLayer_26972394619675
// MI455X (gfx1250) — matrix operand order repaired and checked by reading, not compiled, not run on hardware
//
#include <hip/hip_runtime.h>
#include <hip/hip_bf16.h>
#include <math.h>

typedef __bf16 bf16_t;
typedef __attribute__((ext_vector_type(16))) __bf16 v16bf;
typedef __attribute__((ext_vector_type(8)))  __bf16 v8bf;
typedef __attribute__((ext_vector_type(8)))  float   v8f;
typedef __attribute__((ext_vector_type(4)))  unsigned int uint4v;
typedef __attribute__((ext_vector_type(8)))  int     int8v;
typedef __attribute__((ext_vector_type(4)))  int     int4v;

#if defined(__has_builtin)
#  if __has_builtin(__builtin_amdgcn_tensor_load_to_lds) && __has_builtin(__builtin_amdgcn_s_wait_tensorcnt)
#    define HAVE_TDM 1
#  endif
#endif
#ifndef HAVE_TDM
#  define HAVE_TDM 0
#endif

// ---------------------------------------------------------------------------
// WMMA fragment helpers (CDNA5 bf16 16x16x32, wave32 layouts per ISA 7.12.2)
// A (16x32, MxK): lane L: m=L&15, khi=L>>4. elem e<8 -> K=8*khi+e ; e>=8 -> K=16+8*khi+(e-8)
// B (32x16, KxN): lane L: n=L&15, kg=L>>4.  elem e<8 -> K=8*kg+e ; e>=8 -> K=16+8*kg+(e-8)
// C (16x16 f32):  lane L: n=L&15, mb=8*(L>>4). vgpr v -> M=mb+v
// ---------------------------------------------------------------------------
static __device__ __forceinline__ v16bf load_frag_a(const bf16_t* __restrict__ p, int lda) {
  const int lane = threadIdx.x & 31;
  const int m = lane & 15, khi = lane >> 4;
  const bf16_t* row = p + (size_t)m * lda;
  union { v16bf v; v8bf h[2]; } u;
  u.h[0] = *(const v8bf*)(row + 8 * khi);
  u.h[1] = *(const v8bf*)(row + 16 + 8 * khi);
  return u.v;
}

static __device__ __forceinline__ v16bf load_frag_b(const bf16_t* __restrict__ p, int ldb) {
  const int lane = threadIdx.x & 31;
  const int n = lane & 15, kg = lane >> 4;
  const bf16_t* row = p + (size_t)n * ldb;
  union { v16bf v; v8bf h[2]; } u;
  u.h[0] = *(const v8bf*)(row + 8 * kg);
  u.h[1] = *(const v8bf*)(row + 16 + 8 * kg);
  return u.v;
}

static __device__ __forceinline__ v16bf load_frag_a_lds_f32(const float* __restrict__ p, int lda) {
  const int lane = threadIdx.x & 31;
  const int m = lane & 15, khi = lane >> 4;
  const float* row = p + (size_t)m * lda;
  v16bf r;
  #pragma unroll
  for (int e = 0; e < 8; ++e) r[e] = (__bf16)row[8 * khi + e];
  #pragma unroll
  for (int e = 0; e < 8; ++e) r[8 + e] = (__bf16)row[16 + 8 * khi + e];
  return r;
}

static __device__ __forceinline__ v8f wmma_bf16(v16bf a, v16bf b, v8f c) {
  return __builtin_amdgcn_wmma_f32_16x16x32_bf16(false, a, false, b, (short)0, c, false, false);
}

// ---------------------------------------------------------------------------
// Tensor Data Mover: 2D bf16 tile (tile_k contiguous elems x tile_rows rows,
// row stride = row_stride_elems) from global -> LDS, compacted row-major.
// D# packing per CDNA5 ISA sec 8.3 (group0) / 8.4 (group1); groups 2/3 zero (2D).
// This toolchain's builtin takes 6 args: (g0 u32x4, g1 i32x8, g2 i32x4, g3 i32x4,
// extra i32x8, cpol i32).
// ---------------------------------------------------------------------------
static __device__ __forceinline__ unsigned lds_off(const void* p) {
  return (unsigned)(unsigned long long)(__attribute__((address_space(3))) const char*)p;
}

static __device__ __forceinline__ void tdm_load_2d_bf16(unsigned lds_byte_addr,
                                                        const void* gptr,
                                                        unsigned tile_k,
                                                        unsigned tile_rows,
                                                        unsigned row_stride_elems) {
#if HAVE_TDM
  unsigned long long ga = (unsigned long long)gptr;
  uint4v g0;
  g0[0] = 1u;                                               // count=1, user-mode, no gather
  g0[1] = lds_byte_addr;                                    // lds_addr
  g0[2] = (unsigned)(ga & 0xffffffffu);                     // global_addr[31:0]
  g0[3] = (unsigned)((ga >> 32) & 0x1ffffffu) | (2u << 30); // global_addr[56:32] | type=2
  int8v g1;
  g1[0] = (int)(1u << 16);                                  // data_size=1 (2 bytes), wg_mask=0
  g1[1] = (int)((tile_k & 0xffffu) << 16);                  // tensor_dim0[15:0]
  g1[2] = (int)(((tile_k >> 16) & 0xffffu) | ((tile_rows & 0xffffu) << 16)); // dim0 hi | dim1 lo
  g1[3] = (int)(((tile_rows >> 16) & 0xffffu) | ((tile_k & 0xffffu) << 16)); // dim1 hi | tile_dim0
  g1[4] = (int)(tile_rows & 0xffffu);                       // tile_dim1 | tile_dim2=0
  g1[5] = (int)row_stride_elems;                            // tensor_dim0_stride[31:0]
  g1[6] = 0;
  g1[7] = 0;
  int4v z4 = {0, 0, 0, 0};
  int8v z8 = {0, 0, 0, 0, 0, 0, 0, 0};
  __builtin_amdgcn_tensor_load_to_lds(g0, g1, z4, z4, z8, 0);
#else
  (void)lds_byte_addr; (void)gptr; (void)tile_k; (void)tile_rows; (void)row_stride_elems;
#endif
}

// ---------------------------------------------------------------------------
// Generic WMMA GEMM: C[M,N] = A[M,K] (bf16) x Bt[N,K]^T (bf16) + bias
// Block: 256 threads = 8 waves (4 along M x 2 along N); tile 128x64; wave 32x32.
// B tile (64 rows x 256-deep K chunk) staged into LDS by the Tensor Data Mover,
// double buffered (TENSORcnt), shared by all 8 waves; A fragments stream from
// global (b128 loads). Dynamic LDS: 2 * 64 * 256 * 2B = 64 KB.
// MODE 0: bf16 row-major   MODE 2: bf16 transposed per batch (Vt)   MODE 3: f32 + resid
// ---------------------------------------------------------------------------
template<bool RELU, int MODE>
__global__ __launch_bounds__(256, 2)
void gemm_bf16_kernel(const bf16_t* __restrict__ A, const bf16_t* __restrict__ Bt,
                      const float* __restrict__ bias, void* __restrict__ Cout,
                      const float* __restrict__ resid,
                      int M, int N, int K, int Stok, int Dv) {
  constexpr int KC = 256;
  extern __shared__ bf16_t ldsB[];          // [2][64*KC]
  const int w  = threadIdx.x >> 5;
  const int wm = w & 3;
  const int wn = w >> 2;
  const int row0 = blockIdx.y * 128 + wm * 32;
  const int bcol = blockIdx.x * 64;
  const int lane = threadIdx.x & 31;

  v8f c00 = {}, c01 = {}, c10 = {}, c11 = {};
  const bf16_t* Abase = A  + (size_t)row0 * K;
  const bf16_t* Bblk  = Bt + (size_t)bcol * K;
  const int nch = K / KC;

#if HAVE_TDM
  if (w == 0) tdm_load_2d_bf16(lds_off(ldsB), Bblk, KC, 64, (unsigned)K);
#endif

  for (int c = 0; c < nch; ++c) {
    __syncthreads();                        // readers of buf[(c+1)&1] (iter c-1) are done
#if HAVE_TDM
    if (w == 0) {
      if (c + 1 < nch) {
        tdm_load_2d_bf16(lds_off(ldsB + ((c + 1) & 1) * 64 * KC),
                         Bblk + (size_t)(c + 1) * KC, KC, 64, (unsigned)K);
        __builtin_amdgcn_s_wait_tensorcnt(1);   // chunk c landed (in-order per wave)
      } else {
        __builtin_amdgcn_s_wait_tensorcnt(0);
      }
    }
#else
    {  // cooperative fallback copy of chunk c (only when TDM builtin unavailable)
      bf16_t* dst = ldsB + (c & 1) * 64 * KC;
      for (int i = threadIdx.x; i < 64 * KC / 8; i += 256) {
        int r = i / (KC / 8), q = i % (KC / 8);
        *(v8bf*)(dst + r * KC + q * 8) = *(const v8bf*)(Bblk + (size_t)r * K + (size_t)c * KC + q * 8);
      }
    }
#endif
    __syncthreads();                        // chunk c visible to all waves

    const bf16_t* Bc = ldsB + (c & 1) * 64 * KC + (size_t)(wn * 32) * KC;
    const bf16_t* Ac = Abase + (size_t)c * KC;
    #pragma unroll 4
    for (int kk = 0; kk < KC; kk += 32) {
      if (kk + 64 < KC)                     // stream A ahead (global_prefetch_b8)
        __builtin_prefetch(Ac + (size_t)lane * K + kk + 64, 0, 1);
      v16bf a0 = load_frag_a(Ac + kk, K);
      v16bf a1 = load_frag_a(Ac + (size_t)16 * K + kk, K);
      v16bf b0 = load_frag_b(Bc + kk, KC);
      v16bf b1 = load_frag_b(Bc + (size_t)16 * KC + kk, KC);
      c00 = wmma_bf16(a0, b0, c00);
      c01 = wmma_bf16(a0, b1, c01);
      c10 = wmma_bf16(a1, b0, c10);
      c11 = wmma_bf16(a1, b1, c11);
    }
  }

  const int col0 = bcol + wn * 32;
  const int n  = lane & 15;
  const int mb = (lane >> 4) * 8;
  #pragma unroll
  for (int mi = 0; mi < 2; ++mi) {
    #pragma unroll
    for (int ni = 0; ni < 2; ++ni) {
      v8f c = (mi == 0) ? (ni == 0 ? c00 : c01) : (ni == 0 ? c10 : c11);
      const int col = col0 + ni * 16 + n;
      const float bb = bias ? bias[col] : 0.f;
      if constexpr (MODE == 2) {
        const int m0 = row0 + mi * 16 + mb;     // 8 consecutive tokens, same batch
        const int bi = m0 / Stok;
        const int s  = m0 % Stok;
        v8bf pk;
        #pragma unroll
        for (int v = 0; v < 8; ++v) pk[v] = (__bf16)(c[v] + bb);
        *(v8bf*)((bf16_t*)Cout + ((size_t)bi * Dv + col) * Stok + s) = pk;
      } else {
        #pragma unroll
        for (int v = 0; v < 8; ++v) {
          const int m = row0 + mi * 16 + mb + v;
          float x = c[v] + bb;
          if (RELU) x = fmaxf(x, 0.f);
          if constexpr (MODE == 0)      ((bf16_t*)Cout)[(size_t)m * N + col] = (__bf16)x;
          else if constexpr (MODE == 3) ((float*)Cout)[(size_t)m * N + col]  = x + resid[(size_t)m * N + col];
          else                          ((float*)Cout)[(size_t)m * N + col]  = x;
        }
      }
    }
  }
}

// ---------------------------------------------------------------------------
// Attention: per block one (batch, 32-query tile).
// Phase A: scores(32x2048) = Q @ Kmix^T * scale -> LDS (256 KB of 320 KB WGP LDS)
// Phase B: exact row softmax in LDS (wave32 shuffle reductions)
// Phase C: attn(32x768) = P @ V via Vt (transposed V) WMMA
// ---------------------------------------------------------------------------
__global__ __launch_bounds__(256, 1)
void attn_kernel(const bf16_t* __restrict__ Q, const bf16_t* __restrict__ KM,
                 const bf16_t* __restrict__ Vt, float* __restrict__ attn, float scale) {
  extern __shared__ float sc[];                 // 32 * 2048 floats
  const int S = 2048, D = 768;
  const int b  = blockIdx.y;
  const int q0 = blockIdx.x * 32;
  const int w  = threadIdx.x >> 5;
  const int lane = threadIdx.x & 31;
  const int n  = lane & 15;
  const int mb = (lane >> 4) * 8;

  const bf16_t* Aq = Q + ((size_t)b * S + q0) * D;

  // Phase A: each wave owns 256 score columns (16 n-tiles)
  for (int nt = 0; nt < 16; ++nt) {
    const int nc = w * 256 + nt * 16;
    v8f c0 = {}, c1 = {};
    const bf16_t* Bb = KM + ((size_t)b * S + nc) * D;
    for (int kk = 0; kk < D; kk += 32) {
      v16bf a0 = load_frag_a(Aq + kk, D);
      v16bf a1 = load_frag_a(Aq + (size_t)16 * D + kk, D);
      v16bf bf = load_frag_b(Bb + kk, D);
      c0 = wmma_bf16(a0, bf, c0);
      c1 = wmma_bf16(a1, bf, c1);
    }
    #pragma unroll
    for (int v = 0; v < 8; ++v) {
      sc[(size_t)(mb + v) * S + nc + n]      = c0[v] * scale;
      sc[(size_t)(16 + mb + v) * S + nc + n] = c1[v] * scale;
    }
  }
  __syncthreads();

  // Phase B: each wave softmaxes 4 rows of 2048
  for (int r = 0; r < 4; ++r) {
    float* prow = sc + (size_t)(w * 4 + r) * S;
    float mx = -1e30f;
    for (int j = lane; j < S; j += 32) mx = fmaxf(mx, prow[j]);
    #pragma unroll
    for (int o = 16; o > 0; o >>= 1) mx = fmaxf(mx, __shfl_xor(mx, o, 32));
    float sum = 0.f;
    for (int j = lane; j < S; j += 32) { float e = __expf(prow[j] - mx); prow[j] = e; sum += e; }
    #pragma unroll
    for (int o = 16; o > 0; o >>= 1) sum += __shfl_xor(sum, o, 32);
    const float inv = 1.f / sum;
    for (int j = lane; j < S; j += 32) prow[j] *= inv;
  }
  __syncthreads();

  // Phase C: each wave owns 96 output columns (6 n-tiles), K loop over 2048
  for (int nt = 0; nt < 6; ++nt) {
    const int nc = w * 96 + nt * 16;
    v8f c0 = {}, c1 = {};
    const bf16_t* Bb = Vt + ((size_t)b * D + nc) * S;
    for (int kk = 0; kk < S; kk += 32) {
      v16bf a0 = load_frag_a_lds_f32(sc + kk, S);
      v16bf a1 = load_frag_a_lds_f32(sc + (size_t)16 * S + kk, S);
      v16bf bf = load_frag_b(Bb + kk, S);
      c0 = wmma_bf16(a0, bf, c0);
      c1 = wmma_bf16(a1, bf, c1);
    }
    #pragma unroll
    for (int v = 0; v < 8; ++v) {
      attn[((size_t)b * S + q0 + mb + v) * D + nc + n]      = c0[v];
      attn[((size_t)b * S + q0 + 16 + mb + v) * D + nc + n] = c1[v];
    }
  }
}

// ---------------------------------------------------------------------------
// Small utility kernels
// ---------------------------------------------------------------------------
__global__ void f32_to_bf16_kernel(const float* __restrict__ in, bf16_t* __restrict__ out, long n) {
  long i = (long)blockIdx.x * blockDim.x + threadIdx.x;
  long st = (long)gridDim.x * blockDim.x;
  for (; i < n; i += st) out[i] = (__bf16)in[i];
}

__global__ void copy_f32_kernel(const float* __restrict__ in, float* __restrict__ out, long n) {
  long i = (long)blockIdx.x * blockDim.x + threadIdx.x;
  long st = (long)gridDim.x * blockDim.x;
  for (; i < n; i += st) out[i] = in[i];
}

// Wt[n*K + k] = W[k*N + n]  (weights are tiny; done once)
__global__ void transpose_bf16_kernel(const float* __restrict__ W, bf16_t* __restrict__ Wt, int K, int N) {
  long total = (long)K * N;
  long i = (long)blockIdx.x * blockDim.x + threadIdx.x;
  long st = (long)gridDim.x * blockDim.x;
  for (; i < total; i += st) {
    long nn = i / K, kk = i % K;
    Wt[i] = (__bf16)W[kk * (long)N + nn];
  }
}

// ktot[b][d] = sum_t K[b,t,d]
__global__ void ksum_kernel(const bf16_t* __restrict__ Kb, float* __restrict__ ktot, int S, int D) {
  int d = blockIdx.x * blockDim.x + threadIdx.x;
  int b = blockIdx.y;
  if (d >= D) return;
  const bf16_t* p = Kb + (size_t)b * S * D + d;
  float s = 0.f;
  for (int t = 0; t < S; ++t) s += (float)p[(size_t)t * D];
  ktot[b * D + d] = s;
}

// k_mix[s,d] = (ktot[d] + (e-1)*band_sum(s,d)) / (cnt*e + (S-cnt))
__global__ void kmix_kernel(const bf16_t* __restrict__ Kb, const float* __restrict__ ktot,
                            bf16_t* __restrict__ KM, int S, int D) {
  const int s = blockIdx.x, b = blockIdx.y;
  const int lo = max(s - 2, 0), hi = min(s + 2, S - 1);
  const int cnt = hi - lo + 1;
  const float E = 2.718281828459045f;
  const float denom = cnt * E + (float)(S - cnt);
  const float w_all = 1.f / denom, w_band = (E - 1.f) / denom;
  for (int d = threadIdx.x; d < D; d += blockDim.x) {
    float band = 0.f;
    for (int t = lo; t <= hi; ++t) band += (float)Kb[((size_t)b * S + t) * D + d];
    KM[((size_t)b * S + s) * D + d] = (__bf16)(ktot[b * D + d] * w_all + band * w_band);
  }
}

// Deterministic two-pass global layernorm stats
__global__ void reduce_partials_kernel(const float* __restrict__ a, const float* __restrict__ b,
                                       float* __restrict__ part, long n) {
  __shared__ float ls[256], ls2[256];
  long i = (long)blockIdx.x * blockDim.x + threadIdx.x;
  long st = (long)gridDim.x * blockDim.x;
  float s = 0.f, s2 = 0.f;
  for (; i < n; i += st) {
    float v = a[i] + (b ? b[i] : 0.f);
    s += v; s2 += v * v;
  }
  ls[threadIdx.x] = s; ls2[threadIdx.x] = s2;
  __syncthreads();
  for (int o = 128; o > 0; o >>= 1) {
    if (threadIdx.x < o) { ls[threadIdx.x] += ls[threadIdx.x + o]; ls2[threadIdx.x] += ls2[threadIdx.x + o]; }
    __syncthreads();
  }
  if (threadIdx.x == 0) { part[2 * blockIdx.x] = ls[0]; part[2 * blockIdx.x + 1] = ls2[0]; }
}

__global__ void finalize_stats_kernel(const float* __restrict__ part, int nblk,
                                      float* __restrict__ stats, float invn) {
  __shared__ float ls[256], ls2[256];
  float s = 0.f, s2 = 0.f;
  for (int i = threadIdx.x; i < nblk; i += 256) { s += part[2 * i]; s2 += part[2 * i + 1]; }
  ls[threadIdx.x] = s; ls2[threadIdx.x] = s2;
  __syncthreads();
  for (int o = 128; o > 0; o >>= 1) {
    if (threadIdx.x < o) { ls[threadIdx.x] += ls[threadIdx.x + o]; ls2[threadIdx.x] += ls2[threadIdx.x + o]; }
    __syncthreads();
  }
  if (threadIdx.x == 0) {
    float mean = ls[0] * invn;
    float var  = ls2[0] * invn - mean * mean;
    stats[0] = mean;
    stats[1] = rsqrtf(var + 1e-6f);
  }
}

template<bool OUT_BF16>
__global__ void apply_ln_kernel(const float* __restrict__ a, const float* __restrict__ b,
                                const float* __restrict__ gamma, const float* __restrict__ beta,
                                const float* __restrict__ stats, void* __restrict__ out, long n) {
  const float mean = stats[0], rstd = stats[1];
  long i = (long)blockIdx.x * blockDim.x + threadIdx.x;
  long st = (long)gridDim.x * blockDim.x;
  for (; i < n; i += st) {
    float v = a[i] + (b ? b[i] : 0.f);
    float r = (v - mean) * rstd * gamma[i] + beta[i];
    if (OUT_BF16) ((bf16_t*)out)[i] = (__bf16)r;
    else          ((float*)out)[i]  = r;
  }
}

// ---------------------------------------------------------------------------
extern "C" void kernel_launch(void* const* d_in, const int* in_sizes, int n_in,
                              void* d_out, int out_size, void* d_ws, size_t ws_size,
                              hipStream_t stream) {
  const float* text  = (const float*)d_in[0];
  const float* image = (const float*)d_in[1];
  const float* Wq = (const float*)d_in[2];  const float* bq = (const float*)d_in[3];
  const float* Wk = (const float*)d_in[4];  const float* bk = (const float*)d_in[5];
  const float* Wv = (const float*)d_in[6];  const float* bv = (const float*)d_in[7];
  const float* W1 = (const float*)d_in[8];  const float* b1 = (const float*)d_in[9];
  const float* W2 = (const float*)d_in[10]; const float* b2 = (const float*)d_in[11];
  const float* gamma = (const float*)d_in[12];
  const float* beta  = (const float*)d_in[13];
  float* out = (float*)d_out;

  const int Bsz = 8, S = 2048, Dm = 768, H = 3072;
  const long BS  = (long)Bsz * S;        // 16384
  const long Nel = BS * Dm;              // 12,582,912

  char* base = (char*)d_ws;
  size_t off = 0;
  auto carve = [&](size_t bytes) -> void* {
    void* p = base + off;
    off += (bytes + 255) & ~(size_t)255;
    return p;
  };

  bf16_t* Xb  = (bf16_t*)carve(Nel * 2);
  bf16_t* Qb  = (bf16_t*)carve(Nel * 2);   // Qb,Kb,KM,Vt contiguous -> reused as FFN hidden h
  bf16_t* Kb  = (bf16_t*)carve(Nel * 2);
  bf16_t* KM  = (bf16_t*)carve(Nel * 2);
  bf16_t* Vt  = (bf16_t*)carve(Nel * 2);
  bf16_t* hb  = Qb;                        // [BS, H] bf16 == 4 * Nel bf16
  bf16_t* Wqt = (bf16_t*)carve((size_t)Dm * Dm * 2);
  bf16_t* Wkt = (bf16_t*)carve((size_t)Dm * Dm * 2);
  bf16_t* Wvt = (bf16_t*)carve((size_t)Dm * Dm * 2);
  bf16_t* W1t = (bf16_t*)carve((size_t)H * Dm * 2);
  bf16_t* W2t = (bf16_t*)carve((size_t)Dm * H * 2);
  float*  ktot = (float*)carve((size_t)Bsz * Dm * 4);
  float*  attn = (float*)carve(Nel * 4);   // reused as u = ffn + text
  bf16_t* xln  = (bf16_t*)carve(Nel * 2);
  float*  part = (float*)carve(2048 * 4);
  float*  st1  = (float*)carve(2 * 4);
  float*  st2  = (float*)carve(2 * 4);
  float*  u    = attn;

  // ---- stage inputs / weights in bf16 (weights transposed for B-fragments)
  f32_to_bf16_kernel<<<4096, 256, 0, stream>>>(text, Xb, Nel);
  {
    long ni = (long)Bsz * 196 * Dm;
    copy_f32_kernel<<<2048, 256, 0, stream>>>(image, out + Nel, ni);
  }
  transpose_bf16_kernel<<<1024, 256, 0, stream>>>(Wq, Wqt, Dm, Dm);
  transpose_bf16_kernel<<<1024, 256, 0, stream>>>(Wk, Wkt, Dm, Dm);
  transpose_bf16_kernel<<<1024, 256, 0, stream>>>(Wv, Wvt, Dm, Dm);
  transpose_bf16_kernel<<<4096, 256, 0, stream>>>(W1, W1t, Dm, H);   // W1t[H, Dm]
  transpose_bf16_kernel<<<4096, 256, 0, stream>>>(W2, W2t, H, Dm);   // W2t[Dm, H]

  // ---- allow 64 KB dynamic LDS on the GEMM variants, 256 KB on attention
  const int gemm_smem = 2 * 64 * 256 * 2;
  (void)hipFuncSetAttribute(reinterpret_cast<const void*>(gemm_bf16_kernel<false, 0>),
                            hipFuncAttributeMaxDynamicSharedMemorySize, gemm_smem);
  (void)hipFuncSetAttribute(reinterpret_cast<const void*>(gemm_bf16_kernel<false, 2>),
                            hipFuncAttributeMaxDynamicSharedMemorySize, gemm_smem);
  (void)hipFuncSetAttribute(reinterpret_cast<const void*>(gemm_bf16_kernel<true, 0>),
                            hipFuncAttributeMaxDynamicSharedMemorySize, gemm_smem);
  (void)hipFuncSetAttribute(reinterpret_cast<const void*>(gemm_bf16_kernel<false, 3>),
                            hipFuncAttributeMaxDynamicSharedMemorySize, gemm_smem);

  // ---- QKV projections (WMMA bf16, TDM-staged B tiles)
  dim3 blk(256);
  dim3 gq(Dm / 64, (int)(BS / 128));
  gemm_bf16_kernel<false, 0><<<gq, blk, gemm_smem, stream>>>(Xb, Wqt, bq, Qb, nullptr, (int)BS, Dm, Dm, S, Dm);
  gemm_bf16_kernel<false, 0><<<gq, blk, gemm_smem, stream>>>(Xb, Wkt, bk, Kb, nullptr, (int)BS, Dm, Dm, S, Dm);
  gemm_bf16_kernel<false, 2><<<gq, blk, gemm_smem, stream>>>(Xb, Wvt, bv, Vt, nullptr, (int)BS, Dm, Dm, S, Dm);

  // ---- k_mix = softmax(banded 0/1 mask) @ K, closed form (global + band sums)
  ksum_kernel<<<dim3((Dm + 255) / 256, Bsz), blk, 0, stream>>>(Kb, ktot, S, Dm);
  kmix_kernel<<<dim3(S, Bsz), blk, 0, stream>>>(Kb, ktot, KM, S, Dm);

  // ---- attention: scores -> LDS softmax -> PV (256 KB dynamic LDS per WGP)
  const int smem = 32 * 2048 * 4;
  (void)hipFuncSetAttribute(reinterpret_cast<const void*>(attn_kernel),
                            hipFuncAttributeMaxDynamicSharedMemorySize, smem);
  attn_kernel<<<dim3(S / 32, Bsz), blk, smem, stream>>>(Qb, KM, Vt, attn,
                                                        1.0f / sqrtf((float)Dm));

  // ---- LN1 over ALL elements of (attn + text) -> xln (bf16 for FFN)
  reduce_partials_kernel<<<1024, 256, 0, stream>>>(attn, text, part, Nel);
  finalize_stats_kernel<<<1, 256, 0, stream>>>(part, 1024, st1, 1.0f / (float)Nel);
  apply_ln_kernel<true><<<4096, 256, 0, stream>>>(attn, text, gamma, beta, st1, xln, Nel);

  // ---- FFN (WMMA bf16): h = relu(x@W1+b1); u = h@W2 + b2 + text
  dim3 g1(H / 64, (int)(BS / 128));
  gemm_bf16_kernel<true, 0><<<g1, blk, gemm_smem, stream>>>(xln, W1t, b1, hb, nullptr, (int)BS, H, Dm, S, Dm);
  dim3 g2(Dm / 64, (int)(BS / 128));
  gemm_bf16_kernel<false, 3><<<g2, blk, gemm_smem, stream>>>(hb, W2t, b2, u, text, (int)BS, Dm, H, S, Dm);

  // ---- LN2 over ALL elements of u -> final output (f32)
  reduce_partials_kernel<<<1024, 256, 0, stream>>>(u, nullptr, part, Nel);
  finalize_stats_kernel<<<1, 256, 0, stream>>>(part, 1024, st2, 1.0f / (float)Nel);
  apply_ln_kernel<false><<<4096, 256, 0, stream>>>(u, nullptr, gamma, beta, st2, out, Nel);
}
